// EncoderLayer_42623255445770
// MI455X (gfx1250) — matrix operand order repaired and checked by reading, not compiled, not run on hardware
//
#include <hip/hip_runtime.h>
#include <hip/hip_bf16.h>

typedef __attribute__((ext_vector_type(16))) _Float16 v16h;
typedef __attribute__((ext_vector_type(8)))  _Float16 v8h;
typedef __attribute__((ext_vector_type(8)))  float    v8f;
typedef __attribute__((ext_vector_type(4)))  unsigned int v4u;
typedef __attribute__((ext_vector_type(8)))  int v8i_;
typedef __attribute__((ext_vector_type(4)))  int v4i_;

#define WMMA_F16(a,b,c) __builtin_amdgcn_wmma_f32_16x16x32_f16(false,(a),false,(b),(short)0,(c),false,false)

// ---- problem dims ----
#define BDIM 16
#define NDIM 1024
#define EDIM 256
#define HDIM 16
#define DDIM 16
#define HD   256
#define FFD  512

// A-operand fragment (16x32 f16): lane&15 = row M; lanes 0-15 hold K 0-7 / 16-23,
// lanes 16-31 hold K 8-15 / 24-31. Two contiguous b128 loads.
static __device__ __forceinline__ v16h load_fragA(const _Float16* base, int lane, int k0, int ld) {
  int m = lane & 15, g = (lane >> 4) & 1;
  const _Float16* p = base + (size_t)m * ld + k0 + g * 8;
  v8h lo = *(const v8h*)p;
  v8h hi = *(const v8h*)(p + 16);
  v16h a;
#pragma unroll
  for (int i = 0; i < 8; i++) { a[i] = lo[i]; a[i + 8] = hi[i]; }
  return a;
}

// B-operand fragment (32x16 f16) from row-major B^T (N x K): lane&15 = col N;
// lanes 0-15 hold K 0-7 / 16-23, lanes 16-31 hold K 8-15 / 24-31.
static __device__ __forceinline__ v16h load_fragB(const _Float16* baseCol, int lane, int k0, int ld) {
  int n = lane & 15, g = (lane >> 4) & 1;
  const _Float16* p = baseCol + (size_t)n * ld + k0 + g * 8;
  v8h lo = *(const v8h*)p;
  v8h hi = *(const v8h*)(p + 16);
  v16h b;
#pragma unroll
  for (int i = 0; i < 8; i++) { b[i] = lo[i]; b[i + 8] = hi[i]; }
  return b;
}

// ---------------- WMMA GEMM with TDM-staged B tile ----------------
// C = A(f16,MxK) @ Bt(f16,NxK)^T.  Block tile: 128 rows x 64 cols (8 waves).
// The 64xK B^T slab (contiguous in memory) is DMA'd to LDS once per block by
// the Tensor Data Mover, then all waves stream it via ds_load.
// MODE 0: outH = f16(C * scale)          (q/k projections)
// MODE 1: outH = f16(C), V^T per-head layout [b,h,d,n]   (v projection)
// MODE 2: outF = C + bias + resid(f32)   (Wo + residual, W2 + residual)
// MODE 3: outH = f16(C + bias)           (W1 projection)
template <int MODE, int KT>
__global__ void __launch_bounds__(256) gemm_wmma_k(
    const _Float16* __restrict__ A, const _Float16* __restrict__ Bt,
    const float* __restrict__ bias, const float* __restrict__ resid,
    float* __restrict__ outF, _Float16* __restrict__ outH,
    int M, int Nn, float scale) {
  __shared__ __align__(16) _Float16 btile[64 * KT];   // 32KB (K=256) / 64KB (K=512)

  int lane = threadIdx.x & 31;
  int wid  = threadIdx.x >> 5;
  int tilesX = Nn >> 6;
  int bx = blockIdx.x % tilesX;
  int by = blockIdx.x / tilesX;
  int row0 = by * 128 + wid * 16;
  int col0 = bx * 64;

  // ---- TDM: stage the 64xKT B^T slab into LDS (wave 0 issues, then block barrier) ----
  if (wid == 0) {
    unsigned long long ga = (unsigned long long)(uintptr_t)(Bt + (size_t)col0 * KT);
    unsigned int ldsoff = (unsigned int)(uintptr_t)(&btile[0]);
    const unsigned int numEl = 64u * (unsigned)KT;     // f16 elements (fits 16 bits)
    v4u g0;
    g0[0] = 1u;                                        // count=1, user descriptor
    g0[1] = ldsoff;                                    // lds_addr
    g0[2] = (unsigned int)(ga & 0xffffffffu);          // global_addr[31:0]
    g0[3] = (unsigned int)((ga >> 32) & 0x1ffffffu) | (2u << 30);  // addr[56:32], type=2
    v8i_ g1;
    g1[0] = (int)(1u << 16);                           // data_size = 1 -> 2 bytes/elem
    g1[1] = (int)((numEl & 0xffffu) << 16);            // tensor_dim0[15:0]
    g1[2] = (int)(((numEl >> 16) & 0xffffu) | (1u << 16)); // dim0[31:16], tensor_dim1=1
    g1[3] = (int)((numEl & 0xffffu) << 16);            // dim1[31:16]=0, tile_dim0=numEl
    g1[4] = 1;                                         // tile_dim1=1, tile_dim2=0
    g1[5] = (int)numEl;                                // tensor_dim0_stride[31:0]
    g1[6] = 0;                                         // stride0 hi, stride1 lo
    g1[7] = 0;
    v4i_ g2 = {0, 0, 0, 0};
    v4i_ g3 = {0, 0, 0, 0};
    v8i_ g4 = {0, 0, 0, 0, 0, 0, 0, 0};
    __builtin_amdgcn_tensor_load_to_lds(g0, g1, g2, g3, g4, 0);
    __builtin_amdgcn_s_wait_tensorcnt(0);
  }
  __syncthreads();

  v8f acc[4];
#pragma unroll
  for (int j = 0; j < 4; j++)
#pragma unroll
    for (int r = 0; r < 8; r++) acc[j][r] = 0.f;

  const _Float16* Arow = A + (size_t)row0 * KT;
#pragma unroll
  for (int k0 = 0; k0 < KT; k0 += 32) {
    v16h a = load_fragA(Arow, lane, k0, KT);
#pragma unroll
    for (int j = 0; j < 4; j++) {
      v16h b = load_fragB(&btile[(size_t)(j * 16) * KT], lane, k0, KT);
      acc[j] = WMMA_F16(a, b, acc[j]);
    }
  }

  int g = (lane >> 4) & 1, n = lane & 15;
#pragma unroll
  for (int j = 0; j < 4; j++) {
    int cc = col0 + j * 16 + n;
#pragma unroll
    for (int r = 0; r < 8; r++) {
      int rr = row0 + g * 8 + r;
      float v = acc[j][r];
      if (MODE == 0) {
        outH[(size_t)rr * Nn + cc] = (_Float16)(v * scale);
      } else if (MODE == 1) {
        int bb = rr >> 10, tn = rr & 1023;
        int hcol = cc >> 4, dcol = cc & 15;
        outH[(((size_t)(bb * HDIM + hcol)) * DDIM + dcol) * NDIM + tn] = (_Float16)v;
      } else if (MODE == 2) {
        outF[(size_t)rr * Nn + cc] = v + bias[cc] + resid[(size_t)rr * Nn + cc];
      } else {
        outH[(size_t)rr * Nn + cc] = (_Float16)(v + bias[cc]);
      }
    }
  }
}

// ---------------- flash attention: 1 wave per (b, h, 16-query tile) ----------------
__global__ void __launch_bounds__(256) attn_kernel(
    const _Float16* __restrict__ q, const _Float16* __restrict__ k,
    const _Float16* __restrict__ vT, _Float16* __restrict__ attnh) {
  __shared__ __align__(16) _Float16 pshare[8][16][32];
  int wid = threadIdx.x >> 5, lane = threadIdx.x & 31;
  int gw = blockIdx.x * 8 + wid;                 // 0 .. B*H*(N/16)-1
  int qt = gw & 63;
  int hh = (gw >> 6) & 15;
  int bb = gw >> 10;
  const _Float16* qbase = q  + ((size_t)bb * NDIM) * HD + hh * DDIM;
  const _Float16* kbase = k  + ((size_t)bb * NDIM) * HD + hh * DDIM;
  const _Float16* vbase = vT + ((size_t)(bb * HDIM + hh)) * DDIM * NDIM;
  int m = lane & 15, g = (lane >> 4) & 1;

  // Q fragment: K=0..15 real (D=16), upper half zero (scale 1/sqrt(D) folded into q)
  v16h qa;
  {
    const _Float16* p = qbase + (size_t)(qt * 16 + m) * HD + g * 8;
    v8h lo = *(const v8h*)p;
#pragma unroll
    for (int i = 0; i < 8; i++) { qa[i] = lo[i]; qa[i + 8] = (_Float16)0.f; }
  }

  float rmax[8], rsum[8];
  v8f o;
#pragma unroll
  for (int r = 0; r < 8; r++) { rmax[r] = -1e30f; rsum[r] = 0.f; o[r] = 0.f; }

  for (int kt = 0; kt < NDIM / 32; ++kt) {
    int tok0 = kt * 32;
    // K^T fragments for two 16-key subtiles (K rows 0..15 real, 16..31 zero)
    v16h bk0, bk1;
    {
      const _Float16* p0 = kbase + (size_t)(tok0 + m) * HD + g * 8;
      const _Float16* p1 = kbase + (size_t)(tok0 + 16 + m) * HD + g * 8;
      v8h l0 = *(const v8h*)p0;
      v8h l1 = *(const v8h*)p1;
#pragma unroll
      for (int i = 0; i < 8; i++) { bk0[i] = l0[i]; bk0[i + 8] = (_Float16)0.f; bk1[i] = l1[i]; bk1[i + 8] = (_Float16)0.f; }
    }
    v8f zero;
#pragma unroll
    for (int r = 0; r < 8; r++) zero[r] = 0.f;
    v8f s0 = WMMA_F16(qa, bk0, zero);
    v8f s1 = WMMA_F16(qa, bk1, zero);

    // online softmax over this 32-key strip (rows live in 16-lane halves)
#pragma unroll
    for (int r = 0; r < 8; r++) {
      float tm = fmaxf(s0[r], s1[r]);
#pragma unroll
      for (int msk = 1; msk < 16; msk <<= 1) tm = fmaxf(tm, __shfl_xor(tm, msk));
      float nm = fmaxf(rmax[r], tm);
      float corr = __expf(rmax[r] - nm);
      rmax[r] = nm;
      float p0v = __expf(s0[r] - nm);
      float p1v = __expf(s1[r] - nm);
      float ps = p0v + p1v;
#pragma unroll
      for (int msk = 1; msk < 16; msk <<= 1) ps += __shfl_xor(ps, msk);
      rsum[r] = rsum[r] * corr + ps;
      o[r] *= corr;
      int row = g * 8 + r;
      pshare[wid][row][m]      = (_Float16)p0v;   // D-frag (row, key) -> LDS
      pshare[wid][row][16 + m] = (_Float16)p1v;
    }

    // reread P as A fragment (16 queries x 32 keys)
    v16h pa;
    {
      const _Float16* pp = &pshare[wid][m][g * 8];
      v8h lo = *(const v8h*)pp;
      v8h hi = *(const v8h*)(pp + 16);
#pragma unroll
      for (int i = 0; i < 8; i++) { pa[i] = lo[i]; pa[i + 8] = hi[i]; }
    }
    // V fragment (32 tokens x 16 d) from V^T (d-major, contiguous tokens)
    v16h bv;
    {
      const _Float16* p = vbase + (size_t)m * NDIM + tok0 + g * 8;
      v8h lo = *(const v8h*)p, hi = *(const v8h*)(p + 16);
#pragma unroll
      for (int i = 0; i < 8; i++) { bv[i] = lo[i]; bv[i + 8] = hi[i]; }
    }
    o = WMMA_F16(pa, bv, o);
  }

#pragma unroll
  for (int r = 0; r < 8; r++) {
    float inv = 1.f / rsum[r];
    int row = qt * 16 + g * 8 + r;
    attnh[((size_t)bb * NDIM + row) * HD + hh * DDIM + m] = (_Float16)(o[r] * inv);
  }
}

// ---------------- small helper kernels ----------------
__global__ void transpose_to_h(const float* __restrict__ W, _Float16* __restrict__ Wt,
                               int rows, int cols) {
  int tid = blockIdx.x * blockDim.x + threadIdx.x;
  if (tid >= rows * cols) return;
  int i = tid % rows, o = tid / rows;
  Wt[(size_t)o * rows + i] = (_Float16)W[(size_t)i * cols + o];
}

__global__ void f32_to_h(const float* __restrict__ x, _Float16* __restrict__ xh, int n) {
  int tid = blockIdx.x * blockDim.x + threadIdx.x;
  if (tid < n) xh[tid] = (_Float16)x[tid];
}

__global__ void glu_kernel(const _Float16* __restrict__ proj, _Float16* __restrict__ gbuf, int n) {
  int tid = blockIdx.x * blockDim.x + threadIdx.x;
  if (tid >= n) return;
  int row = tid >> 9, j = tid & 511;
  float xp = (float)proj[(size_t)row * 1024 + j];
  float gt = (float)proj[(size_t)row * 1024 + 512 + j];
  float s = gt / (1.f + __expf(-gt));
  gbuf[tid] = (_Float16)(s * xp);
}

// instance norm over sequence axis (per (b,e) channel); outH optional f16 copy
__global__ void __launch_bounds__(256) instnorm_kernel(
    const float* __restrict__ y, const float* __restrict__ gamma, const float* __restrict__ beta,
    float* __restrict__ outF, _Float16* __restrict__ outH) {
  int b = blockIdx.x, e = threadIdx.x;
  const float* p = y + (size_t)b * NDIM * EDIM + e;
  float s = 0.f, s2 = 0.f;
  for (int n = 0; n < NDIM; n++) { float v = p[(size_t)n * EDIM]; s += v; s2 += v * v; }
  float mean = s * (1.f / NDIM);
  float var = s2 * (1.f / NDIM) - mean * mean;
  float rs = rsqrtf(var + 1e-5f);
  float ga = gamma[e], be = beta[e];
  for (int n = 0; n < NDIM; n++) {
    float v = (p[(size_t)n * EDIM] - mean) * rs * ga + be;
    size_t idx = ((size_t)b * NDIM + n) * EDIM + e;
    outF[idx] = v;
    if (outH) outH[idx] = (_Float16)v;
  }
}

extern "C" void kernel_launch(void* const* d_in, const int* in_sizes, int n_in,
                              void* d_out, int out_size, void* d_ws, size_t ws_size,
                              hipStream_t stream) {
  (void)in_sizes; (void)n_in; (void)out_size; (void)ws_size;
  const float* x      = (const float*)d_in[0];
  const float* Wq     = (const float*)d_in[1];
  const float* Wk     = (const float*)d_in[2];
  const float* Wv     = (const float*)d_in[3];
  const float* Wo     = (const float*)d_in[4];
  const float* bo     = (const float*)d_in[5];
  const float* gamma1 = (const float*)d_in[6];
  const float* beta1  = (const float*)d_in[7];
  const float* gamma2 = (const float*)d_in[8];
  const float* beta2  = (const float*)d_in[9];
  const float* W1     = (const float*)d_in[10];
  const float* b1     = (const float*)d_in[11];
  const float* W2     = (const float*)d_in[12];
  const float* b2     = (const float*)d_in[13];
  float* out = (float*)d_out;
  char* ws = (char*)d_ws;

  // ---- workspace layout (aliased; peak ~82 MB) ----
  _Float16* WqT = (_Float16*)(ws + 0);                      // 256x256 f16
  _Float16* WkT = (_Float16*)(ws + 1 * (1u << 17));
  _Float16* WvT = (_Float16*)(ws + 2 * (1u << 17));
  _Float16* WoT = (_Float16*)(ws + 3 * (1u << 17));
  _Float16* W1T = (_Float16*)(ws + 4 * (1u << 17));         // 1024x256 f16 (512KB)
  _Float16* W2T = (_Float16*)(ws + 4 * (1u << 17) + (1u << 19)); // 256x512 f16 (256KB)
  _Float16* xh    = (_Float16*)(ws + (2u  << 20));          // 8MB
  _Float16* qb    = (_Float16*)(ws + (10u << 20));          // 8MB (pre-scaled 1/sqrt(D))
  _Float16* kb    = (_Float16*)(ws + (18u << 20));          // 8MB
  _Float16* vT    = (_Float16*)(ws + (26u << 20));          // 8MB, [b,h,d,n]
  _Float16* attnh = (_Float16*)(ws + (34u << 20));          // 8MB
  float*    ybuf  = (float*)   (ws + (42u << 20));          // 16MB
  float*    hbuf  = (float*)   (ws + (58u << 20));          // 16MB
  _Float16* hh    = (_Float16*)(ws + (74u << 20));          // 8MB
  _Float16* proj  = (_Float16*)(ws + (2u  << 20));          // 32MB, aliases xh/q/k/v (dead)
  _Float16* gbuf  = (_Float16*)(ws + (34u << 20));          // 16MB, aliases attnh/y-head (dead)
  float*    zbuf  = (float*)   (ws + (2u  << 20));          // 16MB, aliases proj (dead)

  const int M = BDIM * NDIM;  // 16384 rows

  // weight f16 transposes
  transpose_to_h<<<256,  256, 0, stream>>>(Wq, WqT, EDIM, HD);
  transpose_to_h<<<256,  256, 0, stream>>>(Wk, WkT, EDIM, HD);
  transpose_to_h<<<256,  256, 0, stream>>>(Wv, WvT, EDIM, HD);
  transpose_to_h<<<256,  256, 0, stream>>>(Wo, WoT, HD, EDIM);
  transpose_to_h<<<1024, 256, 0, stream>>>(W1, W1T, EDIM, 2 * FFD);
  transpose_to_h<<<512,  256, 0, stream>>>(W2, W2T, FFD, EDIM);
  // x -> f16
  f32_to_h<<<(M * EDIM) / 256, 256, 0, stream>>>(x, xh, M * EDIM);

  // Grid: (M/128) * (Nn/64) blocks of 8 waves (128x64 block tile, TDM-staged B)
  // QKV projections: 128 * 4 = 512 blocks
  gemm_wmma_k<0, EDIM><<<512, 256, 0, stream>>>(xh, WqT, nullptr, nullptr, nullptr, qb, M, HD, 0.25f);
  gemm_wmma_k<0, EDIM><<<512, 256, 0, stream>>>(xh, WkT, nullptr, nullptr, nullptr, kb, M, HD, 1.0f);
  gemm_wmma_k<1, EDIM><<<512, 256, 0, stream>>>(xh, WvT, nullptr, nullptr, nullptr, vT, M, HD, 1.0f);

  // flash attention: B*H*(N/16) = 16384 waves -> 2048 blocks
  attn_kernel<<<2048, 256, 0, stream>>>(qb, kb, vT, attnh);

  // mha = attn @ Wo + bo + x  (f32)
  gemm_wmma_k<2, HD><<<512, 256, 0, stream>>>(attnh, WoT, bo, x, ybuf, nullptr, M, EDIM, 1.0f);
  // h = instnorm(y)
  instnorm_kernel<<<BDIM, 256, 0, stream>>>(ybuf, gamma1, beta1, hbuf, hh);

  // proj = h @ W1 + b1  (f16, 1024 wide): 128 * 16 = 2048 blocks
  gemm_wmma_k<3, EDIM><<<2048, 256, 0, stream>>>(hh, W1T, b1, nullptr, nullptr, proj, M, 2 * FFD, 1.0f);
  // glu: silu(gate) * x_part
  glu_kernel<<<(M * FFD) / 256, 256, 0, stream>>>(proj, gbuf, M * FFD);
  // z = g @ W2 + b2 + h  (f32), K = 512
  gemm_wmma_k<2, FFD><<<512, 256, 0, stream>>>(gbuf, W2T, b2, hbuf, zbuf, nullptr, M, EDIM, 1.0f);
  // out = instnorm(z)
  instnorm_kernel<<<BDIM, 256, 0, stream>>>(zbuf, gamma2, beta2, out, nullptr);
}
